// BahdanauAttention_87832081203828
// MI455X (gfx1250) — compile-verified, matrix operand order checked by reading, not run on hardware
//
#include <hip/hip_runtime.h>

// ---- problem constants (from reference setup) ----
#define B_  32
#define T_  2048
#define E_  1024
#define A_  512
#define SCALING 2.0f

typedef __attribute__((ext_vector_type(16))) __bf16        bf16x16;
typedef __attribute__((ext_vector_type(8)))  float         f32x8;
typedef __attribute__((ext_vector_type(4)))  float         f32x4;
typedef __attribute__((ext_vector_type(4)))  unsigned int  u32x4;

union BVec {
    unsigned short u[16];
    u32x4          q[2];
    bf16x16        v;
};

// Hardware tanh: CDNA5 has V_TANH_F32 in the TRANS unit (co-executes with
// XDL/WMMA per ISA §8.4). libm tanhf expands to ~20 VALU ops; this is 1 op.
__device__ __forceinline__ float fast_tanh(float x) {
#if __has_builtin(__builtin_amdgcn_tanhf)
    return __builtin_amdgcn_tanhf(x);
#elif __has_builtin(__builtin_amdgcn_tanh_f32)
    return __builtin_amdgcn_tanh_f32(x);
#elif defined(__gfx1250__)
    float r;
    asm volatile("v_tanh_f32 %0, %1" : "=v"(r) : "v"(x));
    return r;
#else
    return tanhf(x);
#endif
}

// ---------------------------------------------------------------------------
// 1) W_enc (E x A, f32, row-major) -> Wt (A x E, bf16)   [1 MB, L2-resident]
// ---------------------------------------------------------------------------
__global__ __launch_bounds__(256) void wt_kernel(const float* __restrict__ W,
                                                 unsigned short* __restrict__ Wt) {
    int idx = blockIdx.x * 256 + threadIdx.x;          // < A_*E_
    int a = idx >> 10;                                 // /E_
    int e = idx & (E_ - 1);
    union { __bf16 h; unsigned short u; } cv;
    cv.h = (__bf16)W[(size_t)e * A_ + a];
    Wt[idx] = cv.u;                                    // Wt[a*E_ + e]
}

// ---------------------------------------------------------------------------
// 2) dec[b][a] = b_enc[a] + sum_d h[b][d] * W_dec[d][a]      (tiny GEMV)
// ---------------------------------------------------------------------------
__global__ __launch_bounds__(512) void dec_kernel(const float* __restrict__ h,
                                                  const float* __restrict__ Wd,
                                                  const float* __restrict__ be,
                                                  float* __restrict__ dec) {
    int b = blockIdx.x;
    int a = threadIdx.x;                               // 512 threads
    float s = be[a];
    for (int d = 0; d < E_; ++d)
        s += h[b * E_ + d] * Wd[(size_t)d * A_ + a];   // coalesced over a
    dec[b * A_ + a] = s;
}

// ---------------------------------------------------------------------------
// 3) Fused energy kernel:  e[b][t] = sum_a tanh(enc@W_enc + dec) * g[a]
//    bf16 WMMA 16x16x32, f32 accumulate. 8 waves x 16 rows = 128 t-rows/WG.
// ---------------------------------------------------------------------------
__global__ __launch_bounds__(256) void e_kernel(const float* __restrict__ enc,
                                                const unsigned short* __restrict__ Wt,
                                                const float* __restrict__ dec,
                                                const float* __restrict__ g,
                                                float* __restrict__ e_out) {
    __shared__ unsigned short Bs[A_ * 32];             // 32 KB: 512 n-rows x 32 k
    const int b    = blockIdx.y;
    const int tid  = threadIdx.x;
    const int wave = tid >> 5;
    const int lane = tid & 31;
    const int half = lane >> 4;                        // 0: lanes 0-15, 1: 16-31
    const int mrow = lane & 15;
    const int t0   = blockIdx.x * 128 + wave * 16;

    const float* arow = enc + (size_t)(b * T_ + t0 + mrow) * E_;

    float accE[8];
    #pragma unroll
    for (int j = 0; j < 8; ++j) accE[j] = 0.0f;

    for (int npass = 0; npass < 2; ++npass) {
        f32x8 acc[16];
        #pragma unroll
        for (int nt = 0; nt < 16; ++nt) acc[nt] = (f32x8)0.0f;

        for (int kk = 0; kk < 32; ++kk) {
            const int k0 = kk * 32;
            __syncthreads();
            // cooperative stage of W tile: Bs[n][0..31] = Wt[n][k0..k0+31]
            #pragma unroll
            for (int r = 0; r < 2; ++r) {
                int n = tid + r * 256;
                const u32x4* src = (const u32x4*)(Wt + (size_t)n * E_ + k0);
                u32x4* dst = (u32x4*)(Bs + n * 32);
                #pragma unroll
                for (int q = 0; q < 4; ++q) dst[q] = src[q];
            }
            __syncthreads();

            // A operand: fp32 loads -> bf16, per §7.12.2 16-bit A layout
            const f32x4* ap  = (const f32x4*)(arow + k0 + half * 8);
            const f32x4* ap2 = (const f32x4*)(arow + k0 + 16 + half * 8);
            f32x4 a0 = ap[0],  a1 = ap[1];
            f32x4 a2 = ap2[0], a3 = ap2[1];
            float af[16] = {a0.x,a0.y,a0.z,a0.w, a1.x,a1.y,a1.z,a1.w,
                            a2.x,a2.y,a2.z,a2.w, a3.x,a3.y,a3.z,a3.w};
            BVec av;
            #pragma unroll
            for (int i = 0; i < 16; ++i) av.v[i] = (__bf16)af[i];

            #pragma unroll
            for (int nt = 0; nt < 16; ++nt) {
                int n = npass * 256 + nt * 16 + mrow;
                const u32x4* bp = (const u32x4*)(Bs + n * 32 + half * 8);
                BVec bv;
                bv.q[0] = bp[0];                       // K = 8*half .. +8
                bv.q[1] = bp[2];                       // K = 16+8*half .. +8
                acc[nt] = __builtin_amdgcn_wmma_f32_16x16x32_bf16(
                    false, av.v, false, bv.v, (short)0, acc[nt], false, false);
            }
        }

        // fused epilogue: tanh(pre + dec) * g, accumulated per lane
        #pragma unroll
        for (int nt = 0; nt < 16; ++nt) {
            int n = npass * 256 + nt * 16 + mrow;
            float dv = dec[b * A_ + n];
            float gv = g[n];
            #pragma unroll
            for (int j = 0; j < 8; ++j)
                accE[j] += fast_tanh(acc[nt][j] + dv) * gv;
        }
    }

    // reduce over the 16 lanes of each half (n dimension), then write e
    #pragma unroll
    for (int j = 0; j < 8; ++j) {
        float v = accE[j];
        #pragma unroll
        for (int m = 8; m >= 1; m >>= 1) v += __shfl_xor(v, m, 32);
        if (mrow == 0)
            e_out[b * T_ + t0 + half * 8 + j] = v;     // row M = j (+8 for hi half)
    }
}

// ---------------------------------------------------------------------------
// 4) masked, scaled softmax over T per batch row -> w (written into d_out)
// ---------------------------------------------------------------------------
__global__ __launch_bounds__(256) void softmax_kernel(const float* __restrict__ e_ws,
                                                      const long long* __restrict__ lens,
                                                      const float* __restrict__ gb,
                                                      float* __restrict__ w_out) {
    __shared__ float red[256];
    int b = blockIdx.x, tid = threadIdx.x;
    long long L = lens[b];
    int len = (int)(L < 1 ? 1 : L);
    float gbv = gb[0];

    float lmax = -3.4e38f;
    for (int t = tid; t < T_; t += 256)
        if (t < len) lmax = fmaxf(lmax, SCALING * (e_ws[b * T_ + t] + gbv));
    red[tid] = lmax; __syncthreads();
    for (int s = 128; s > 0; s >>= 1) {
        if (tid < s) red[tid] = fmaxf(red[tid], red[tid + s]);
        __syncthreads();
    }
    float gmax = red[0]; __syncthreads();

    float lsum = 0.0f;
    for (int t = tid; t < T_; t += 256)
        if (t < len) lsum += expf(SCALING * (e_ws[b * T_ + t] + gbv) - gmax);
    red[tid] = lsum; __syncthreads();
    for (int s = 128; s > 0; s >>= 1) {
        if (tid < s) red[tid] += red[tid + s];
        __syncthreads();
    }
    float inv = 1.0f / red[0];

    for (int t = tid; t < T_; t += 256) {
        float wv = (t < len) ? expf(SCALING * (e_ws[b * T_ + t] + gbv) - gmax) * inv
                             : 0.0f;
        w_out[b * T_ + t] = wv;
    }
}

// ---------------------------------------------------------------------------
// 5) context: c[b][e] = sum_t w[b][t] * enc[b][t][e]  (two-stage, no atomics)
// ---------------------------------------------------------------------------
__global__ __launch_bounds__(256) void ctx_partial(const float* __restrict__ enc,
                                                   const float* __restrict__ w,
                                                   float* __restrict__ part) {
    int b = blockIdx.z, chunk = blockIdx.y;
    int e = blockIdx.x * 256 + threadIdx.x;
    int tbeg = chunk * 128;
    float s = 0.0f;
    for (int t = tbeg; t < tbeg + 128; ++t)
        s += w[b * T_ + t] * enc[(size_t)(b * T_ + t) * E_ + e];
    part[(size_t)(b * 16 + chunk) * E_ + e] = s;
}

__global__ __launch_bounds__(256) void ctx_reduce(const float* __restrict__ part,
                                                  float* __restrict__ c) {
    int i = blockIdx.x * 256 + threadIdx.x;            // < B_*E_
    int b = i >> 10, e = i & (E_ - 1);
    float s = 0.0f;
    #pragma unroll
    for (int ch = 0; ch < 16; ++ch)
        s += part[(size_t)(b * 16 + ch) * E_ + e];
    c[i] = s;
}

// ---------------------------------------------------------------------------
extern "C" void kernel_launch(void* const* d_in, const int* in_sizes, int n_in,
                              void* d_out, int out_size, void* d_ws, size_t ws_size,
                              hipStream_t stream) {
    const float*      enc   = (const float*)d_in[0];       // (B,T,E) f32
    const long long*  lens  = (const long long*)d_in[1];   // (B,) i64
    const float*      hdec  = (const float*)d_in[2];       // (B,D) f32
    const float*      W_enc = (const float*)d_in[3];       // (E,A) f32
    const float*      b_enc = (const float*)d_in[4];       // (A,) f32
    const float*      W_dec = (const float*)d_in[5];       // (D,A) f32
    const float*      gvecw = (const float*)d_in[6];       // (A,1) f32
    const float*      gvecb = (const float*)d_in[7];       // (1,) f32

    float* c_out = (float*)d_out;                          // (B,E)
    float* w_out = (float*)d_out + B_ * E_;                // (B,T)

    char* ws = (char*)d_ws;
    unsigned short* Wt   = (unsigned short*)(ws);                    // 1 MB
    float*          decb = (float*)(ws + 1048576);                   // 64 KB
    float*          ebuf = (float*)(ws + 1048576 + 65536);           // 256 KB
    float*          part = (float*)(ws + 1048576 + 65536 + 262144);  // 2 MB

    wt_kernel<<<(A_ * E_) / 256, 256, 0, stream>>>(W_enc, Wt);
    dec_kernel<<<B_, 512, 0, stream>>>(hdec, W_dec, b_enc, decb);
    e_kernel<<<dim3(T_ / 128, B_), 256, 0, stream>>>(enc, Wt, decb, gvecw, ebuf);
    softmax_kernel<<<B_, 256, 0, stream>>>(ebuf, lens, gvecb, w_out);
    ctx_partial<<<dim3(E_ / 256, 16, B_), 256, 0, stream>>>(enc, w_out, part);
    ctx_reduce<<<(B_ * E_) / 256, 256, 0, stream>>>(part, c_out);
}
